// UniMatchLSTM_28346784153923
// MI455X (gfx1250) — matrix operand order repaired and checked by reading, compiled, not run on hardware
//
#include <hip/hip_runtime.h>

// ---------------------------------------------------------------------------
// MI455X (gfx1250) Match-LSTM.
//   * Hoist Hq@Wq.T (+bq) and Hp[t]@Wp.T (+bp) into parallel bf16 WMMA GEMMs.
//   * Sequential 384-step scan as a 64-workgroup cooperative kernel:
//       S1  u[b,:]   = P[t,b,:] + br + h @ Wr.T           (block-local, f32 FMA)
//       S2  logits -> softmax -> q_alpha[b,:]             (block-local, v_exp/v_rcp)
//       --- grid barrier ---
//       S5a gates = [hp|q_alpha|h] @ [Wih|Whh].T + bias   (WMMA, 256 tiles/step)
//       --- grid barrier ---
//       S5b LSTM pointwise -> c, h (f32 + swizzled bf16)  + write out[t]
//       --- grid barrier ---
//   * A/B matrices pre-swizzled to per-lane WMMA layouts (K = e + 16*half); consecutive
//     K-tiles are contiguous so inner loops are base-pointer + immediate-offset
//     b128 loads (no per-tile VALU address math) feeding v_wmma.
// ---------------------------------------------------------------------------

typedef __bf16 v16bf __attribute__((ext_vector_type(16)));
typedef float  v8f   __attribute__((ext_vector_type(8)));

#define NWG  64
#define NTHR 256

__device__ __forceinline__ unsigned short f2bf(float f) {
  union { float f; unsigned int u; } c; c.f = f;
  unsigned int r = c.u + 0x7fffu + ((c.u >> 16) & 1u);   // round-to-nearest-even
  return (unsigned short)(r >> 16);
}
__device__ __forceinline__ float fast_tanh(float x) {
  float e = __builtin_amdgcn_exp2f(x * 2.8853900817779268f);   // e^{2x}
  return 1.0f - 2.0f * __builtin_amdgcn_rcpf(e + 1.0f);
}
__device__ __forceinline__ float fast_sigmoid(float x) {
  float e = __builtin_amdgcn_exp2f(x * -1.4426950408889634f);
  return __builtin_amdgcn_rcpf(1.0f + e);
}
__device__ __forceinline__ float fast_exp(float x) {
  return __builtin_amdgcn_exp2f(x * 1.4426950408889634f);
}
__device__ __forceinline__ v16bf ldtile(const unsigned short* p) {
  return *reinterpret_cast<const v16bf*>(p);
}

// ---- layout helpers -------------------------------------------------------
// A-tile (16x32, 16-bit) per-lane element e -> K, contiguous (matches B):
//   lanes 0-15 : K = 0..15 ; lanes 16-31 : K = 16..31
__device__ __forceinline__ int a_kof(int e, int half) {
  return e + half * 16;
}
// inverse: K-local (0..31) -> (half, e)
__device__ __forceinline__ void a_inv(int kl, int& half, int& e) {
  half = kl >> 4;
  e    = kl & 15;
}

// ---- swizzle f32 [R,K] row-major -> bf16 tiles [(rt*KT+kt)][lane][e] ------
// bmode=0: A layout (K = e + 16*half, same as B). bmode=1: B layout (K = e + 16*half).
__global__ void uml_swizzle(const float* __restrict__ src,
                            unsigned short* __restrict__ dst,
                            int KT, int K, int total, int bmode) {
  for (int d = blockIdx.x * blockDim.x + threadIdx.x; d < total;
       d += gridDim.x * blockDim.x) {
    int e    = d & 15;
    int lane = (d >> 4) & 31;
    int tile = d >> 9;
    int rt   = tile / KT;
    int kt   = tile - rt * KT;
    int half = lane >> 4;
    int r    = rt * 16 + (lane & 15);
    int k    = kt * 32 + (bmode ? (e + half * 16) : a_kof(e, half));
    dst[d] = f2bf(src[(size_t)r * K + k]);
  }
}

__global__ void uml_transpose256(const float* __restrict__ src,
                                 float* __restrict__ dst) {
  int i = blockIdx.x * blockDim.x + threadIdx.x;   // 65536 elems
  if (i < 65536) { int r = i >> 8, c = i & 255; dst[c * 256 + r] = src[i]; }
}

// ---- pre-swizzled bf16 WMMA GEMM (K fixed at 256 = 8 k-tiles):
//      C[m,n] = sum_k A[m,k]*W[n,k] + bias[n]
__global__ void uml_gemm_bf16(const unsigned short* __restrict__ Asw,
                              const unsigned short* __restrict__ Bsw,
                              const float* __restrict__ bias,
                              float* __restrict__ C,
                              int MT, int NT, int ldc) {
  int lane = threadIdx.x & 31;
  int gw   = (blockIdx.x * blockDim.x + threadIdx.x) >> 5;
  int nw   = (gridDim.x * blockDim.x) >> 5;
  int half = lane >> 4;
  for (int tile = gw; tile < MT * NT; tile += nw) {
    int mt = tile / NT, nt = tile - mt * NT;
    const unsigned short* ap = Asw + (((size_t)mt << 3) << 9) + lane * 16;
    const unsigned short* bp = Bsw + (((size_t)nt << 3) << 9) + lane * 16;
    v8f acc = {0.f, 0.f, 0.f, 0.f, 0.f, 0.f, 0.f, 0.f};
#pragma unroll
    for (int kt = 0; kt < 8; ++kt) {
      v16bf a = ldtile(ap + kt * 512);   // contiguous k-tiles -> imm offsets
      v16bf b = ldtile(bp + kt * 512);
      acc = __builtin_amdgcn_wmma_f32_16x16x32_bf16(false, a, false, b,
                                                    (short)0, acc, false, false);
    }
    int   n  = nt * 16 + (lane & 15);
    float bn = bias[n];
#pragma unroll
    for (int v = 0; v < 8; ++v) {
      int m = mt * 16 + v + 8 * half;
      C[(size_t)m * ldc + n] = acc[v] + bn;
    }
  }
}

// ---- state / barrier init (runs every call -> deterministic replays) ------
__global__ void uml_init(float* h, float* c, unsigned short* h_sw,
                         unsigned int* bar) {
  int i = blockIdx.x * blockDim.x + threadIdx.x;
  if (i < 16384) { h[i] = 0.f; c[i] = 0.f; h_sw[i] = 0; }
  if (i == 0) { bar[0] = 0u; bar[1] = 0u; }
}

// ---- agent-scope grid barrier (64 resident blocks) ------------------------
__device__ __forceinline__ void grid_barrier(unsigned int* bar) {
  __threadfence();
  __syncthreads();
  if (threadIdx.x == 0) {
    unsigned int g = __hip_atomic_load(&bar[1], __ATOMIC_RELAXED,
                                       __HIP_MEMORY_SCOPE_AGENT);
    unsigned int a = __hip_atomic_fetch_add(&bar[0], 1u, __ATOMIC_ACQ_REL,
                                            __HIP_MEMORY_SCOPE_AGENT);
    if (a == NWG - 1) {
      __hip_atomic_store(&bar[0], 0u, __ATOMIC_RELAXED,
                         __HIP_MEMORY_SCOPE_AGENT);
      __hip_atomic_fetch_add(&bar[1], 1u, __ATOMIC_RELEASE,
                             __HIP_MEMORY_SCOPE_AGENT);
    } else {
      while (__hip_atomic_load(&bar[1], __ATOMIC_ACQUIRE,
                               __HIP_MEMORY_SCOPE_AGENT) == g)
        __builtin_amdgcn_s_sleep(1);
    }
  }
  __syncthreads();
}

// ---- the sequential scan (64 blocks x 256 threads, block b owns batch b) --
__global__ __launch_bounds__(NTHR, 1)
void uml_scan(const float* __restrict__ Hq,
              const unsigned char* __restrict__ mask,
              const float* __restrict__ WrT,   // [K=256][H=256] f32
              const float* __restrict__ br,
              const float* __restrict__ Wg,
              const float* __restrict__ bg,
              const float* __restrict__ bih,
              const float* __restrict__ bhh,
              const unsigned short* __restrict__ WihB,  // B-swiz [64nt][16kt]
              const unsigned short* __restrict__ WhhB,  // B-swiz [64nt][8kt]
              const unsigned short* __restrict__ HpA,   // A-swiz [1536mt][8kt]
              const float* __restrict__ wq_hq,          // [Q*B,256] f32
              const float* __restrict__ P,              // [T*B,256] f32
              float* __restrict__ h_f32,                // [64,256]
              unsigned short* __restrict__ h_sw,        // A-swiz [4mt][8kt]
              unsigned short* __restrict__ qa_sw,       // A-swiz [4mt][8kt]
              float* __restrict__ c_st,                 // [64,256]
              float* __restrict__ gates,                // [64,1024]
              unsigned int* __restrict__ bar,
              float* __restrict__ out)                  // [384,64,256]
{
  const int tid  = threadIdx.x;
  const int lane = tid & 31;
  const int wave = tid >> 5;
  const int b    = blockIdx.x;          // owned batch row

  __shared__ float u_s[256];
  __shared__ float part_s[256];
  __shared__ float alpha_s[64];

  // hoisted per-wave tile bases for S5a (only t-dependent part varies)
  const int gw  = b * 8 + wave;         // 0..511 global wave id
  const int mt  = gw >> 6;              // batch tile       (valid when gw<256)
  const int nt  = gw & 63;              // column tile of 4H=1024
  const unsigned short* wp0 = WihB + (((size_t)nt << 4) << 9) + lane * 16; // kt 0..7
  const unsigned short* wp1 = wp0 + 8 * 512;                               // kt 8..15
  const unsigned short* wp2 = WhhB + (((size_t)nt << 3) << 9) + lane * 16;
  const unsigned short* ap1 = qa_sw + (((size_t)mt << 3) << 9) + lane * 16;
  const unsigned short* ap2 = h_sw  + (((size_t)mt << 3) << 9) + lane * 16;

  for (int t = 0; t < 384; ++t) {
    // ---- S1: u[h] = P[t,b,h] + br[h] + sum_k h[b,k] * Wr[h,k] (block-local)
    {
      float acc = P[((size_t)t * 64 + b) * 256 + tid] + br[tid];
      const float* hb = h_f32 + b * 256;
      for (int k = 0; k < 256; ++k)
        acc += hb[k] * WrT[k * 256 + tid];
      u_s[tid] = acc;
    }
    __syncthreads();

    // ---- S2: logits (tanh reduction), softmax, q_alpha (block-local)
    {
      int q = tid >> 2, quar = tid & 3;
      const float* wrow = wq_hq + ((size_t)(q * 64 + b)) * 256 + quar * 64;
      const float* up   = u_s + quar * 64;
      const float* wg   = Wg + quar * 64;
      float acc = 0.f;
      for (int hh = 0; hh < 64; ++hh)
        acc += wg[hh] * fast_tanh(wrow[hh] + up[hh]);
      part_s[tid] = acc;
    }
    __syncthreads();
    if (tid < 64) {
      float l = part_s[4 * tid] + part_s[4 * tid + 1] +
                part_s[4 * tid + 2] + part_s[4 * tid + 3] + bg[0];
      if (!mask[b * 64 + tid]) l = -1e9f;
      alpha_s[tid] = l;
    }
    __syncthreads();
    if (tid == 0) {
      float mx = -3.4e38f;
      for (int q = 0; q < 64; ++q) mx = fmaxf(mx, alpha_s[q]);
      float s = 0.f;
      for (int q = 0; q < 64; ++q) {
        float e = fast_exp(alpha_s[q] - mx); alpha_s[q] = e; s += e;
      }
      float inv = __builtin_amdgcn_rcpf(s);
      for (int q = 0; q < 64; ++q) alpha_s[q] *= inv;
    }
    __syncthreads();
    {
      // q_alpha[b, i=tid] -> write directly into A-swizzled bf16 layout
      float acc = 0.f;
      for (int q = 0; q < 64; ++q)
        acc += alpha_s[q] * Hq[((size_t)(q * 64 + b)) * 256 + tid];
      int amt = b >> 4, kt = tid >> 5, kl = tid & 31, hf, e;
      a_inv(kl, hf, e);
      int ln = hf * 16 + (b & 15);
      qa_sw[(((amt * 8 + kt) << 5) + ln) * 16 + e] = f2bf(acc);
    }
    grid_barrier(bar);

    // ---- S5a: gates GEMM via WMMA. 256 tiles (4 b-tiles x 64 col-tiles)
    if (gw < 256) {
      const int half = lane >> 4;
      v8f acc = {0.f, 0.f, 0.f, 0.f, 0.f, 0.f, 0.f, 0.f};
      const unsigned short* ap0 =
          HpA + ((((size_t)t * 4 + mt) * 8) << 9) + lane * 16;
#pragma unroll
      for (int kt = 0; kt < 8; ++kt) {            // K 0..255 : hp_t part
        v16bf a = ldtile(ap0 + kt * 512);
        v16bf w = ldtile(wp0 + kt * 512);
        acc = __builtin_amdgcn_wmma_f32_16x16x32_bf16(false, a, false, w,
                                                      (short)0, acc, false, false);
      }
#pragma unroll
      for (int kt = 0; kt < 8; ++kt) {            // K 256..511 : q_alpha part
        v16bf a = ldtile(ap1 + kt * 512);
        v16bf w = ldtile(wp1 + kt * 512);
        acc = __builtin_amdgcn_wmma_f32_16x16x32_bf16(false, a, false, w,
                                                      (short)0, acc, false, false);
      }
#pragma unroll
      for (int kt = 0; kt < 8; ++kt) {            // recurrent h part (Whh)
        v16bf a = ldtile(ap2 + kt * 512);
        v16bf w = ldtile(wp2 + kt * 512);
        acc = __builtin_amdgcn_wmma_f32_16x16x32_bf16(false, a, false, w,
                                                      (short)0, acc, false, false);
      }
      int   n  = nt * 16 + (lane & 15);
      float bn = bih[n] + bhh[n];
#pragma unroll
      for (int v = 0; v < 8; ++v) {
        int m = mt * 16 + v + 8 * half;
        gates[(size_t)m * 1024 + n] = acc[v] + bn;
      }
    }
    grid_barrier(bar);

    // ---- S5b: LSTM pointwise, write c / h (f32 + swizzled bf16) / out[t]
    {
      int idx = b * 256 + tid;          // covers 0..16383 across grid
      int bb = idx >> 8, hh = idx & 255;
      const float* g = gates + (size_t)bb * 1024;
      float ig = fast_sigmoid(g[hh]);
      float fg = fast_sigmoid(g[256 + hh]);
      float gg = fast_tanh(g[512 + hh]);
      float og = fast_sigmoid(g[768 + hh]);
      float cn = fg * c_st[idx] + ig * gg;
      float hn = og * fast_tanh(cn);
      c_st[idx]  = cn;
      h_f32[idx] = hn;
      out[(size_t)t * 16384 + idx] = hn;
      int amt = bb >> 4, kt = hh >> 5, kl = hh & 31, hf, e;
      a_inv(kl, hf, e);
      int ln = hf * 16 + (bb & 15);
      h_sw[(((amt * 8 + kt) << 5) + ln) * 16 + e] = f2bf(hn);
    }
    grid_barrier(bar);
  }
}

// ---------------------------------------------------------------------------
extern "C" void kernel_launch(void* const* d_in, const int* in_sizes, int n_in,
                              void* d_out, int out_size, void* d_ws, size_t ws_size,
                              hipStream_t stream) {
  (void)in_sizes; (void)n_in; (void)out_size; (void)ws_size;
  const float* Hp  = (const float*)d_in[0];
  const float* Hq  = (const float*)d_in[1];
  const unsigned char* Hq_mask = (const unsigned char*)d_in[2];
  const float* Wq  = (const float*)d_in[3];
  const float* bq  = (const float*)d_in[4];
  const float* Wp  = (const float*)d_in[5];
  const float* bp  = (const float*)d_in[6];
  const float* Wr  = (const float*)d_in[7];
  const float* br  = (const float*)d_in[8];
  const float* Wg  = (const float*)d_in[9];
  const float* bg  = (const float*)d_in[10];
  const float* Wih = (const float*)d_in[11];
  const float* Whh = (const float*)d_in[12];
  const float* bih = (const float*)d_in[13];
  const float* bhh = (const float*)d_in[14];

  char* ws = (char*)d_ws;
  size_t o = 0;
  unsigned short* WqB   = (unsigned short*)(ws + o); o += (size_t)65536  * 2;
  unsigned short* WpB   = (unsigned short*)(ws + o); o += (size_t)65536  * 2;
  unsigned short* WihB  = (unsigned short*)(ws + o); o += (size_t)524288 * 2;
  unsigned short* WhhB  = (unsigned short*)(ws + o); o += (size_t)262144 * 2;
  float*          WrT   = (float*)(ws + o);          o += (size_t)65536  * 4;
  unsigned short* HpA   = (unsigned short*)(ws + o); o += (size_t)6291456 * 2;
  unsigned short* HqA   = (unsigned short*)(ws + o); o += (size_t)1048576 * 2;
  float*          wq_hq = (float*)(ws + o);          o += (size_t)1048576 * 4;
  float*          Pm    = (float*)(ws + o);          o += (size_t)6291456 * 4;
  float*          h_f32 = (float*)(ws + o);          o += (size_t)16384 * 4;
  unsigned short* h_sw  = (unsigned short*)(ws + o); o += (size_t)16384 * 2;
  unsigned short* qa_sw = (unsigned short*)(ws + o); o += (size_t)16384 * 2;
  float*          c_st  = (float*)(ws + o);          o += (size_t)16384 * 4;
  float*          gates = (float*)(ws + o);          o += (size_t)65536 * 4;
  unsigned int*   bar   = (unsigned int*)(ws + o);   o += 256;   // ~46.6 MB total

  // --- weight / activation pre-swizzle (bf16 tile layouts) ---
  uml_swizzle<<<64,   256, 0, stream>>>(Wq,  WqB,  8,  256, 65536,   1);
  uml_swizzle<<<64,   256, 0, stream>>>(Wp,  WpB,  8,  256, 65536,   1);
  uml_swizzle<<<256,  256, 0, stream>>>(Wih, WihB, 16, 512, 524288,  1);
  uml_swizzle<<<128,  256, 0, stream>>>(Whh, WhhB, 8,  256, 262144,  1);
  uml_transpose256<<<256, 256, 0, stream>>>(Wr, WrT);
  uml_swizzle<<<1024, 256, 0, stream>>>(Hp,  HpA,  8,  256, 6291456, 0);
  uml_swizzle<<<256,  256, 0, stream>>>(Hq,  HqA,  8,  256, 1048576, 0);

  // --- hoisted parallel WMMA GEMMs: wq_hq = Hq@Wq.T+bq ; P[t] = Hp[t]@Wp.T+bp
  uml_gemm_bf16<<<128, 256, 0, stream>>>(HqA, WqB, bq, wq_hq, 256,  16, 256);
  uml_gemm_bf16<<<512, 256, 0, stream>>>(HpA, WpB, bp, Pm,    1536, 16, 256);

  // --- state + barrier init, then the cooperative sequential scan ---
  uml_init<<<64, 256, 0, stream>>>(h_f32, c_st, h_sw, bar);
  uml_scan<<<NWG, NTHR, 0, stream>>>(Hq, Hq_mask, WrT, br, Wg, bg, bih, bhh,
                                     WihB, WhhB, HpA, wq_hq, Pm,
                                     h_f32, h_sw, qa_sw, c_st, gates, bar,
                                     (float*)d_out);
}
